// SSDLoss_42863773614777
// MI455X (gfx1250) — compile-verified, matrix operand order checked by reading, not run on hardware
//
#include <hip/hip_runtime.h>
#include <hip/hip_bf16.h>

// Problem constants (match reference)
#define Bc 16
#define Pc 32768
#define Gc 32
#define Cc 81
#define KPAD 96           // 81 padded to 3 * 32 for WMMA K-chunks
#define TILES_PER_IMG (Pc / 16)

typedef __attribute__((ext_vector_type(16))) _Float16 v16h;
typedef __attribute__((ext_vector_type(8)))  float    v8f;
typedef __attribute__((ext_vector_type(4)))  unsigned v4u;
typedef __attribute__((ext_vector_type(8)))  int      v8i;
typedef __attribute__((ext_vector_type(4)))  int      v4i;

__device__ __forceinline__ float wave_fsum(float v) {
  #pragma unroll
  for (int o = 16; o > 0; o >>= 1) v += __shfl_xor(v, o, 32);
  return v;
}
__device__ __forceinline__ float wave_fmax(float v) {
  #pragma unroll
  for (int o = 16; o > 0; o >>= 1) v = fmaxf(v, __shfl_xor(v, o, 32));
  return v;
}
__device__ __forceinline__ float smooth_l1(float d) {
  float ad = fabsf(d);
  return (ad < 1.0f) ? 0.5f * d * d : ad - 0.5f;
}

// ---------------------------------------------------------------------------
// Workspace layout:
//   acc[0] = sum over matched pairs of (conf @ onehot) dot products
//   acc[1] = sum over rows of count_row * (rowmax + log(sumexp))
//   acc[2] = loc loss sum
//   acc[3 + b] = per-image hard-negative loss
//   total[b]   = per-image matched-pair count (int)
//   negv[b*P+p]  = matched-row ? -1 : softmax-max (= 1/sumexp)
//   masks[b*P+p] = 32-bit match mask vs the 32 gt boxes
// ---------------------------------------------------------------------------

__global__ void k_init(float* acc, int* total) {
  int i = threadIdx.x;
  if (i < 3 + Bc) acc[i] = 0.0f;
  if (i < Bc)     total[i] = 0;
}

// One wave per prediction row. 8 rows / block (256 threads).
__global__ __launch_bounds__(256)
void k_rowstats(const float* __restrict__ pred_boxes,
                const float* __restrict__ conf,
                const float* __restrict__ gt_boxes,
                float* __restrict__ acc, int* __restrict__ total,
                float* __restrict__ negv, unsigned* __restrict__ masks)
{
  __shared__ float4 sgt[Gc];
  const int wave = threadIdx.x >> 5;
  const int lane = threadIdx.x & 31;
  const long long rowg = (long long)blockIdx.x * 8 + wave;   // global row
  const int b = (int)(rowg >> 15);                           // P = 2^15
  const int p = (int)(rowg & (Pc - 1));

  if (threadIdx.x < Gc) {
    const float4* g4 = (const float4*)(gt_boxes + (size_t)b * Gc * 4);
    sgt[threadIdx.x] = g4[threadIdx.x];
  }
  __syncthreads();

  // IoU of this row's box vs gt box 'lane'
  const float4 pb = ((const float4*)pred_boxes)[(size_t)b * Pc + p];
  const float4 gb = sgt[lane];
  float lx = fmaxf(pb.x, gb.x), ly = fmaxf(pb.y, gb.y);
  float rx = fminf(pb.z, gb.z), ry = fminf(pb.w, gb.w);
  float inter = fmaxf(rx - lx, 0.0f) * fmaxf(ry - ly, 0.0f);
  float aa = fmaxf(pb.z - pb.x, 0.0f) * fmaxf(pb.w - pb.y, 0.0f);
  float ab = fmaxf(gb.z - gb.x, 0.0f) * fmaxf(gb.w - gb.y, 0.0f);
  float uni = fmaxf(aa + ab - inter, 1e-9f);
  bool matched = (inter / uni) > 0.5f;
  unsigned mask = (unsigned)__ballot(matched);
  int cnt = __popc(mask);

  // smooth-L1 (mean over 4 coords) where matched
  float sl = 0.25f * (smooth_l1(pb.x - gb.x) + smooth_l1(pb.y - gb.y) +
                      smooth_l1(pb.z - gb.z) + smooth_l1(pb.w - gb.w));
  float locp = wave_fsum(matched ? sl : 0.0f);

  // row softmax stats over 81 logits: lanes cover {lane, lane+32, lane+64}
  const float* crow = conf + (size_t)rowg * Cc;
  float x0 = crow[lane];
  float x1 = crow[lane + 32];
  float x2 = (lane < Cc - 64) ? crow[lane + 64] : -__builtin_inff();
  float m = wave_fmax(fmaxf(x0, fmaxf(x1, x2)));
  float e = expf(x0 - m) + expf(x1 - m) + ((lane < Cc - 64) ? expf(x2 - m) : 0.0f);
  float s = wave_fsum(e);

  if (lane == 0) {
    negv[rowg]  = mask ? -1.0f : (1.0f / s);   // softmax max = exp(0)/s
    masks[rowg] = mask;
    atomicAdd(&acc[2], locp);
    atomicAdd(&acc[1], (float)cnt * (m + logf(s)));
    atomicAdd(&total[b], cnt);
  }
}

// WMMA one-hot GEMM: per wave, one 16-pred x 32-gt tile of (conf @ onehot).
// A-tile staged by the Tensor Data Mover (zero-padded 81->96 via tensor OOB
// semantics); B one-hot prebuilt in WMMA fragment layout for b128 LDS reads.
__global__ __launch_bounds__(256)
void k_gemm(const float* __restrict__ conf,
            const int* __restrict__ gt_labels,
            const unsigned* __restrict__ masks,
            float* __restrict__ acc)
{
  __shared__ __attribute__((aligned(16))) float    ldsRaw[8][16 * KPAD];
  __shared__ __attribute__((aligned(32))) _Float16 bfrag[3][2][32][16];
  __shared__ int slbl[Gc];
  const int wave = threadIdx.x >> 5;
  const int lane = threadIdx.x & 31;
  const long long t = (long long)blockIdx.x * 8 + wave;
  const int b    = (int)(t / TILES_PER_IMG);
  const int row0 = (int)(t % TILES_PER_IMG) * 16;

  // ---- A tile: TDM load of 16 rows x 81 f32, tile width 96 so columns
  //      81..95 come back as zeros (OOB reads of the tensor return zero). ----
#if __has_builtin(__builtin_amdgcn_tensor_load_to_lds)
  {
    unsigned ldsOff = (unsigned)(unsigned long long)(uintptr_t)&ldsRaw[wave][0];
    unsigned long long ga =
        (unsigned long long)(uintptr_t)(conf + ((size_t)b * Pc + row0) * Cc);
    // D# group0: count=1 | lds_addr | global_addr | type=2
    v4u g0 = { 1u, ldsOff, (unsigned)ga, (unsigned)(ga >> 32) | (2u << 30) };
    // D# group1: data_size=4B; tensor_dim0=81; tensor_dim1=16;
    //            tile_dim0=96; tile_dim1=16; tensor_dim0_stride=81
    v8i g1 = { (int)(2u << 16),
               (int)((unsigned)Cc << 16),
               (int)(16u << 16),
               (int)((unsigned)KPAD << 16),
               16,
               Cc,
               0, 0 };
    v4i z4 = {0, 0, 0, 0};
#if defined(__clang_major__) && __clang_major__ >= 23
    v8i z8 = {0, 0, 0, 0, 0, 0, 0, 0};
    __builtin_amdgcn_tensor_load_to_lds(g0, g1, z4, z4, z8, 0);
#else
    __builtin_amdgcn_tensor_load_to_lds(g0, g1, z4, z4, 0);
#endif
  }
#else
  for (int idx = lane; idx < 16 * KPAD; idx += 32) {
    int r = idx / KPAD, c = idx % KPAD;
    ldsRaw[wave][idx] =
        (c < Cc) ? conf[((size_t)b * Pc + row0 + r) * Cc + c] : 0.0f;
  }
#endif

  if (threadIdx.x < Gc) slbl[threadIdx.x] = gt_labels[b * Gc + threadIdx.x];
  __syncthreads();

  // ---- one-hot B directly in CDNA5 16-bit B-fragment layout:
  //      lane<16 holds K {0..7,16..23}+kc, lane>=16 adds +8; N = lane&15. ----
  for (int i = threadIdx.x; i < 3 * 2 * 32 * 8; i += 256) {
    int hp = i & 7;                 // pair of packed halves (h = 2hp, 2hp+1)
    int ln = (i >> 3) & 31;
    int nt = (i >> 8) & 1;
    int ch = i >> 9;
    int col = nt * 16 + (ln & 15);
    int lbl = slbl[col];
    int kb = ((hp < 4) ? 2 * hp : 16 + 2 * (hp - 4)) +
             ((ln & 16) ? 8 : 0) + ch * 32;
    bfrag[ch][nt][ln][2 * hp]     = (lbl == kb)     ? (_Float16)1.0f : (_Float16)0.0f;
    bfrag[ch][nt][ln][2 * hp + 1] = (lbl == kb + 1) ? (_Float16)1.0f : (_Float16)0.0f;
  }
  __syncthreads();
#if __has_builtin(__builtin_amdgcn_tensor_load_to_lds) && \
    __has_builtin(__builtin_amdgcn_s_wait_tensorcnt)
  __builtin_amdgcn_s_wait_tensorcnt(0);
#endif
  __asm__ volatile("" ::: "memory");

  const int row  = lane & 15;
  const int hiK  = (lane & 16) ? 8 : 0;
  const int roff = (lane & 16) ? 8 : 0;

  // one mask load per lane; broadcast rows via shuffle later
  unsigned mymask = masks[(size_t)b * Pc + row0 + (lane & 15)];

  v8f c0 = {}, c1 = {};
  const float4* rawrow = (const float4*)&ldsRaw[wave][row * KPAD];
  #pragma unroll
  for (int ch = 0; ch < 3; ++ch) {
    int kq = (ch * 32 + hiK) >> 2;          // float4 index of K-group base
    float4 f0 = rawrow[kq];                 // k' = base+0..3   (h 0..3)
    float4 f1 = rawrow[kq + 1];             // k' = base+4..7   (h 4..7)
    float4 f2 = rawrow[kq + 4];             // k' = base+16..19 (h 8..11)
    float4 f3 = rawrow[kq + 5];             // k' = base+20..23 (h 12..15)
    v16h a;
    a[0]  = (_Float16)f0.x; a[1]  = (_Float16)f0.y;
    a[2]  = (_Float16)f0.z; a[3]  = (_Float16)f0.w;
    a[4]  = (_Float16)f1.x; a[5]  = (_Float16)f1.y;
    a[6]  = (_Float16)f1.z; a[7]  = (_Float16)f1.w;
    a[8]  = (_Float16)f2.x; a[9]  = (_Float16)f2.y;
    a[10] = (_Float16)f2.z; a[11] = (_Float16)f2.w;
    a[12] = (_Float16)f3.x; a[13] = (_Float16)f3.y;
    a[14] = (_Float16)f3.z; a[15] = (_Float16)f3.w;
    v16h b0 = *(const v16h*)&bfrag[ch][0][lane][0];
    v16h b1 = *(const v16h*)&bfrag[ch][1][lane][0];
    c0 = __builtin_amdgcn_wmma_f32_16x16x32_f16(false, a, false, b0,
                                                (short)0, c0, false, false);
    c1 = __builtin_amdgcn_wmma_f32_16x16x32_f16(false, a, false, b1,
                                                (short)0, c1, false, false);
  }

  // mask by match matrix and reduce (C/D: elem r <-> M = r + roff, N = lane&15)
  float sum = 0.0f;
  const int g0c = lane & 15, g1c = 16 + (lane & 15);
  #pragma unroll
  for (int r = 0; r < 8; ++r) {
    unsigned mk = (unsigned)__shfl((int)mymask, r + roff, 32);
    if ((mk >> g0c) & 1u) sum += c0[r];
    if ((mk >> g1c) & 1u) sum += c1[r];
  }
  sum = wave_fsum(sum);
  if (lane == 0) atomicAdd(&acc[0], sum);
}

// Exact top-k sum of log1p over non-negative negv, per image.
// 3-level radix-histogram selection on float bit patterns (order-preserving
// for non-negative floats). k = 3 * total[b]. Ties handled by multiplicity.
__global__ __launch_bounds__(1024)
void k_topk(const float* __restrict__ negv, const int* __restrict__ total,
            float* __restrict__ acc)
{
  const int b = blockIdx.x;
  const float* nv = negv + (size_t)b * Pc;
  __shared__ unsigned hist[4096];
  __shared__ int sh_t, sh_r, sh_all;
  __shared__ float rbuf[32];

  const long long k = 3LL * (long long)total[b];
  float mysum = 0.0f, extra = 0.0f;

  if (k > 0) {
    // level 0: bits >> 20  (values in (0,1] -> bin <= 0x3F8)
    for (int i = threadIdx.x; i < 4096; i += 1024) hist[i] = 0;
    __syncthreads();
    for (int i = threadIdx.x; i < Pc; i += 1024) {
      float v = nv[i];
      if (v >= 0.0f) atomicAdd(&hist[__float_as_uint(v) >> 20], 1u);
    }
    __syncthreads();
    if (threadIdx.x == 0) {
      long long cum = 0; int t = 4095;
      for (; t >= 0; --t) { cum += hist[t]; if (cum >= k) break; }
      if (t < 0) { sh_all = 1; sh_t = 0; sh_r = 0; }
      else { sh_all = 0; sh_t = t; sh_r = (int)(k - (cum - (long long)hist[t])); }
    }
    __syncthreads();

    if (sh_all) {                       // fewer than k non-negatives: keep all
      for (int i = threadIdx.x; i < Pc; i += 1024) {
        float v = nv[i];
        if (v >= 0.0f) mysum += log1pf(v);
      }
    } else {
      const unsigned t0 = (unsigned)sh_t;
      const int r0 = sh_r;
      __syncthreads();
      // level 1: (bits >> 8) & 0xFFF within bin t0
      for (int i = threadIdx.x; i < 4096; i += 1024) hist[i] = 0;
      __syncthreads();
      for (int i = threadIdx.x; i < Pc; i += 1024) {
        float v = nv[i];
        if (v >= 0.0f) {
          unsigned u = __float_as_uint(v);
          if ((u >> 20) == t0) atomicAdd(&hist[(u >> 8) & 0xFFFu], 1u);
        }
      }
      __syncthreads();
      if (threadIdx.x == 0) {
        long long cum = 0; int t = 4095;
        for (; t >= 0; --t) { cum += hist[t]; if (cum >= r0) break; }
        sh_t = (t < 0) ? 0 : t;
        sh_r = (t < 0) ? 0 : (int)(r0 - (cum - (long long)hist[t]));
      }
      __syncthreads();
      const unsigned t1 = (unsigned)sh_t;
      const int r1 = sh_r;
      __syncthreads();
      // level 2: bits & 0xFF within (t0,t1)
      for (int i = threadIdx.x; i < 256; i += 1024) hist[i] = 0;
      __syncthreads();
      for (int i = threadIdx.x; i < Pc; i += 1024) {
        float v = nv[i];
        if (v >= 0.0f) {
          unsigned u = __float_as_uint(v);
          if ((u >> 8) == ((t0 << 12) | t1)) atomicAdd(&hist[u & 0xFFu], 1u);
        }
      }
      __syncthreads();
      if (threadIdx.x == 0) {
        long long cum = 0; int t = 255;
        for (; t >= 0; --t) { cum += hist[t]; if (cum >= r1) break; }
        sh_t = (t < 0) ? 0 : t;
        sh_r = (t < 0) ? 0 : (int)(r1 - (cum - (long long)hist[t]));
      }
      __syncthreads();
      const unsigned t2 = (unsigned)sh_t;
      const int r2 = sh_r;
      const unsigned ubits = (t0 << 20) | (t1 << 8) | t2;  // k-th largest bits
      for (int i = threadIdx.x; i < Pc; i += 1024) {
        float v = nv[i];
        if (v >= 0.0f && __float_as_uint(v) > ubits) mysum += log1pf(v);
      }
      if (threadIdx.x == 0)
        extra = (float)r2 * log1pf(__uint_as_float(ubits));
    }
  }

  float w = wave_fsum(mysum);
  const int wave = threadIdx.x >> 5, lane = threadIdx.x & 31;
  if (lane == 0) rbuf[wave] = w;
  __syncthreads();
  if (threadIdx.x < 32) {
    float v = rbuf[threadIdx.x];
    v = wave_fsum(v);
    if (threadIdx.x == 0) acc[3 + b] = v + extra;
  }
}

__global__ void k_final(const float* __restrict__ acc,
                        const int* __restrict__ total,
                        float* __restrict__ out)
{
  const int lane = threadIdx.x;
  float nm = (lane < Bc) ? acc[3 + lane] : 0.0f;
  float nsum = wave_fsum(nm);
  if (lane == 0) {
    float matches_loss = acc[0] - acc[1];       // sum matched logsm gathered
    float conf_loss = -matches_loss + nsum;
    out[0] = (conf_loss + acc[2]) / (float)total[Bc - 1];
  }
}

extern "C" void kernel_launch(void* const* d_in, const int* in_sizes, int n_in,
                              void* d_out, int out_size, void* d_ws, size_t ws_size,
                              hipStream_t stream) {
  const float* pred_boxes = (const float*)d_in[0];   // B*P*4
  const float* conf       = (const float*)d_in[1];   // B*P*C
  const float* gt_boxes   = (const float*)d_in[2];   // B*G*4
  const int*   gt_labels  = (const int*)d_in[3];     // B*G

  float*    acc   = (float*)d_ws;                    // 3 + B floats
  int*      total = (int*)(acc + 3 + Bc);            // B ints
  float*    negv  = (float*)(total + Bc);            // B*P floats
  unsigned* masks = (unsigned*)(negv + (size_t)Bc * Pc); // B*P uints
  float*    out   = (float*)d_out;

  k_init<<<1, 64, 0, stream>>>(acc, total);
  k_rowstats<<<(Bc * Pc) / 8, 256, 0, stream>>>(pred_boxes, conf, gt_boxes,
                                                acc, total, negv, masks);
  k_gemm<<<(Bc * TILES_PER_IMG) / 8, 256, 0, stream>>>(conf, gt_labels,
                                                       masks, acc);
  k_topk<<<Bc, 1024, 0, stream>>>(negv, total, acc);
  k_final<<<1, 32, 0, stream>>>(acc, total, out);
}
